// TTTLayer_42606075576726
// MI455X (gfx1250) — compile-verified, matrix operand order checked by reading, not run on hardware
//
#include <hip/hip_runtime.h>
#include <hip/hip_bf16.h>
#include <math.h>

// ---------------- problem constants ----------------
constexpr int BATCH = 4;
constexpr int SEQ   = 4096;
constexpr int DIM   = 2048;   // D
constexpr int HID   = 512;    // TTT
constexpr int MBK   = 16;     // minibatch chunk
constexpr int NCH   = 256;    // SEQ / MBK
constexpr int NSPL  = 8;      // d-splits of W per batch in the scan
constexpr float LRC = 0.1f;
constexpr float EPS = 1e-5f;

// ---------------- CDNA5 types ----------------
typedef __attribute__((ext_vector_type(16))) __bf16 v16bf;
typedef __attribute__((ext_vector_type(8)))  float  v8f;
typedef __attribute__((ext_vector_type(4)))  unsigned int v4u;
typedef __attribute__((ext_vector_type(8)))  int v8i;
typedef __attribute__((ext_vector_type(4)))  int v4i;

#define WMMA_BF16(A, Bm, C) \
  __builtin_amdgcn_wmma_f32_16x16x32_bf16(false, (A), false, (Bm), (short)0, (C), false, false)

__device__ __forceinline__ unsigned short f2bf(float f) {
  union { float f; unsigned u; } v; v.f = f;
  unsigned r = v.u + 0x7FFFu + ((v.u >> 16) & 1u);   // RTNE
  return (unsigned short)(r >> 16);
}
__device__ __forceinline__ float bf2f(unsigned short b) {
  union { unsigned u; float f; } v; v.u = ((unsigned)b) << 16;
  return v.f;
}

// Load a 16x32-bf16 WMMA operand half-row for this lane.
// Lane L holds row (L&15); elements 0..7 = K kb..kb+7, 8..15 = K kb+16..kb+23,
// kb = (L>>4)*8. rowptr must already point at (row, k0+kb). 16B aligned.
__device__ __forceinline__ v16bf load_tile_bf(const unsigned short* rowptr) {
  union { uint4 q[2]; v16bf v; } u;
  u.q[0] = *(const uint4*)(rowptr);
  u.q[1] = *(const uint4*)(rowptr + 16);
  return u.v;
}
// Same but source is fp32 (recurrent W state): convert 16 values to bf16.
__device__ __forceinline__ v16bf load_tile_f32_to_bf(const float* rowptr) {
  unsigned short t[16];
#pragma unroll
  for (int i = 0; i < 8; ++i) t[i]     = f2bf(rowptr[i]);
#pragma unroll
  for (int i = 0; i < 8; ++i) t[8 + i] = f2bf(rowptr[16 + i]);
  v16bf v; __builtin_memcpy(&v, t, 32);
  return v;
}

// ============ K0: conversions + state init + counter zeroing ============
__global__ __launch_bounds__(256) void k0_setup(
    const float* __restrict__ x,  const float* __restrict__ Win,
    const float* __restrict__ Wtg, const float* __restrict__ Wout,
    const float* __restrict__ Winit, const float* __restrict__ binit,
    unsigned short* __restrict__ xbf, unsigned short* __restrict__ wzt,
    unsigned short* __restrict__ woutbf,
    float* __restrict__ Wst, float* __restrict__ bst,
    int* __restrict__ gcnt, float* __restrict__ gpart) {
  const size_t stride = (size_t)gridDim.x * blockDim.x;
  const size_t t0 = (size_t)blockIdx.x * blockDim.x + threadIdx.x;
  for (size_t i = t0; i < (size_t)BATCH * SEQ * DIM; i += stride) xbf[i] = f2bf(x[i]);
  for (size_t i = t0; i < (size_t)2 * HID * DIM; i += stride) {
    size_t r = i / DIM, c = i % DIM;
    wzt[i] = f2bf(r < HID ? Win[r * DIM + c] : Wtg[(r - HID) * DIM + c]);
  }
  for (size_t i = t0; i < (size_t)DIM * HID; i += stride) woutbf[i] = f2bf(Wout[i]);
  for (size_t i = t0; i < (size_t)BATCH * HID * HID; i += stride) Wst[i] = Winit[i % (HID * HID)];
  for (size_t i = t0; i < (size_t)BATCH * HID; i += stride) bst[i] = binit[i % HID];
  for (size_t i = t0; i < (size_t)BATCH * NCH; i += stride) gcnt[i] = 0;
  for (size_t i = t0; i < (size_t)BATCH * NCH * NSPL; i += stride) gpart[i] = 0.f;
}

// ============ K1: z||target projection, [16384,2048] x [2048,1024]^T ============
// One wave = one 16-row x 64-col strip (4 accumulators), software-pipelined so
// next K-step's tiles are in flight while the current WMMAs execute.
__global__ __launch_bounds__(256) void k1_proj(
    const unsigned short* __restrict__ xbf, const unsigned short* __restrict__ wzt,
    const float* __restrict__ b_in, const float* __restrict__ b_tgt,
    unsigned short* __restrict__ zbf, float* __restrict__ tgt) {
  const int wave  = blockIdx.x * (blockDim.x >> 5) + (threadIdx.x >> 5);
  const int lane  = threadIdx.x & 31;
  const int mtile = wave >> 4;        // 1024 m-tiles
  const int strip = wave & 15;        // 16 strips of 64 columns
  const int nl  = lane & 15;
  const int kb  = (lane >> 4) << 3;
  const int mhi = (lane >> 4) << 3;
  const int n0  = strip << 6;

  const unsigned short* aptr = xbf + (size_t)((mtile << 4) + nl) * DIM + kb;
  const unsigned short* b0 = wzt + (size_t)(n0 + nl)      * DIM + kb;
  const unsigned short* b1 = wzt + (size_t)(n0 + 16 + nl) * DIM + kb;
  const unsigned short* b2 = wzt + (size_t)(n0 + 32 + nl) * DIM + kb;
  const unsigned short* b3 = wzt + (size_t)(n0 + 48 + nl) * DIM + kb;

  v8f acc0 = {}, acc1 = {}, acc2 = {}, acc3 = {};
  v16bf a   = load_tile_bf(aptr);
  v16bf tb0 = load_tile_bf(b0);
  v16bf tb1 = load_tile_bf(b1);
  v16bf tb2 = load_tile_bf(b2);
  v16bf tb3 = load_tile_bf(b3);
  for (int k0 = 32; k0 < DIM; k0 += 32) {
    v16bf an  = load_tile_bf(aptr + k0);
    v16bf bn0 = load_tile_bf(b0 + k0);
    v16bf bn1 = load_tile_bf(b1 + k0);
    v16bf bn2 = load_tile_bf(b2 + k0);
    v16bf bn3 = load_tile_bf(b3 + k0);
    __builtin_prefetch(aptr + k0 + 96, 0, 1);   // global_prefetch_b8
    __builtin_prefetch(b0   + k0 + 96, 0, 1);
    acc0 = WMMA_BF16(a, tb0, acc0);
    acc1 = WMMA_BF16(a, tb1, acc1);
    acc2 = WMMA_BF16(a, tb2, acc2);
    acc3 = WMMA_BF16(a, tb3, acc3);
    a = an; tb0 = bn0; tb1 = bn1; tb2 = bn2; tb3 = bn3;
  }
  acc0 = WMMA_BF16(a, tb0, acc0);
  acc1 = WMMA_BF16(a, tb1, acc1);
  acc2 = WMMA_BF16(a, tb2, acc2);
  acc3 = WMMA_BF16(a, tb3, acc3);

#pragma unroll
  for (int r = 0; r < 8; ++r) {
    const int mg = (mtile << 4) + mhi + r;
    v8f accs[4] = {acc0, acc1, acc2, acc3};
#pragma unroll
    for (int j = 0; j < 4; ++j) {
      const int ng = n0 + j * 16 + nl;
      if (ng < HID) {
        float v = accs[j][r] + b_in[ng];
        zbf[(size_t)mg * HID + ng] = f2bf(v);
      } else {
        float v = accs[j][r] + b_tgt[ng - HID];
        tgt[(size_t)mg * HID + (ng - HID)] = v;
      }
    }
  }
}

// ============ K2: sequential TTT scan ============
// grid = BATCH*NSPL blocks; block (b,s) exclusively owns 64 rows of W_b.
// Chunk activations are staged into LDS by the Tensor Data Mover (one
// tensor_load_to_lds per step, issued by wave 5, padded rows in hardware),
// then: pred strip via WMMA (W rows -> bf16 on the fly), CC^T Gram via WMMA
// (A==B registers), gnorm via tr((EE^T)(CC^T)) with deterministic cross-block
// partial exchange, then fp32 rank-16 W update.
__global__ __launch_bounds__(256) void k2_scan(
    const unsigned short* __restrict__ zbf, const float* __restrict__ tgt,
    float* __restrict__ Wst, float* __restrict__ bst,
    unsigned short* __restrict__ preds,
    int* __restrict__ gcnt, float* __restrict__ gpart) {
  __shared__ unsigned short cz[16][HID + 8];   // chunk activations (bf16), 16B row pad
  __shared__ float Esh[16][64 + 4];            // this block's error slab
  __shared__ float G2[16][16];                 // C C^T Gram
  __shared__ float red[256];

  const int b  = blockIdx.x / NSPL;
  const int s  = blockIdx.x % NSPL;
  const int d0 = s * 64;
  const int tid  = threadIdx.x;
  const int wv   = tid >> 5;
  const int lane = tid & 31;
  const int nl  = lane & 15;
  const int kb  = (lane >> 4) << 3;
  const int mhi = (lane >> 4) << 3;

  float* wblk = Wst + ((size_t)b * HID + d0) * HID;
  float* bvec = bst + (size_t)b * HID;

  for (int step = 0; step < NCH; ++step) {
    const unsigned short* zrow = zbf + ((size_t)b * SEQ + (size_t)step * MBK) * HID;
    __syncthreads();                      // previous step done with cz/Esh

    if (wv == 5) {
      // ---- TDM: DMA the 16x512 bf16 tile into LDS with hardware row padding.
      // D# group0: count=1 | lds_addr | 57b global_addr | type=2 ("image")
      const unsigned long long ga = (unsigned long long)(uintptr_t)zrow;
      const unsigned lds = (unsigned)(uintptr_t)&cz[0][0];  // flat[31:0] = LDS offset
      v4u g0;
      g0.x = 1u;                                            // count=1 (valid user D#)
      g0.y = lds;
      g0.z = (unsigned)(ga & 0xFFFFFFFFu);
      g0.w = (unsigned)((ga >> 32) & 0x01FFFFFFu) | (2u << 30);
      // D# group1: data_size=2B, pad_enable, pad_interval=256dw (one 1KB row),
      // pad_amount=4dw (16B) -> LDS row stride = HID+8 bf16. tensor_dim0=512,
      // tensor_dim1=16, tile=512x16, dim0_stride=512.
      v8i g1;
      g1[0] = (int)((1u << 16) | (1u << 20) | (7u << 22) | (3u << 25)); // 0x07D10000
      g1[1] = (int)(512u << 16);   // tensor_dim0[15:0] @ bits 63:48
      g1[2] = (int)(16u << 16);    // tensor_dim1[15:0] @ bits 95:80
      g1[3] = (int)(512u << 16);   // tile_dim0 @ bits 127:112
      g1[4] = 16;                  // tile_dim1 @ bits 143:128
      g1[5] = 512;                 // tensor_dim0_stride[31:0]
      g1[6] = 0;
      g1[7] = 0;
      const v4i gz4 = {0, 0, 0, 0};                 // groups 2/3 unused (2D tensor)
      const v8i gz8 = {0, 0, 0, 0, 0, 0, 0, 0};     // extended group (clang-23 form)
      __builtin_amdgcn_tensor_load_to_lds(g0, g1, gz4, gz4, gz8, 0);
      __builtin_amdgcn_s_wait_tensorcnt(0);
    }
    __syncthreads();                      // publish TDM-written LDS to all waves

    if (wv < 4) {
      // pred tile: rows m x cols d0+wv*16..+16, K = 512 over e
      v8f acc = {};
      const int dg = d0 + wv * 16 + nl;
      const float* wrow = Wst + ((size_t)b * HID + dg) * HID + kb;
      const unsigned short* arow = &cz[nl][kb];
      for (int k0 = 0; k0 < HID; k0 += 32) {
        union { uint4 q[2]; v16bf v; } ua;
        ua.q[0] = *(const uint4*)(arow + k0);         // ds_load_b128
        ua.q[1] = *(const uint4*)(arow + k0 + 16);
        acc = WMMA_BF16(ua.v, load_tile_f32_to_bf(wrow + k0), acc);
      }
      const float bb = bvec[dg];
#pragma unroll
      for (int r = 0; r < 8; ++r) {
        const int m = mhi + r;
        const float pv = acc[r] + bb;                 // pred BEFORE update -> output
        const size_t ro = ((size_t)b * SEQ + (size_t)step * MBK + m) * HID + dg;
        Esh[m][dg - d0] = pv - tgt[ro];
        preds[ro] = f2bf(pv);
      }
    } else if (wv == 4) {
      // G2 = C C^T : A and B operands are the same lane data (row nl of cz)
      v8f acc = {};
      const unsigned short* arow = &cz[nl][kb];
      for (int k0 = 0; k0 < HID; k0 += 32) {
        union { uint4 q[2]; v16bf v; } ua;
        ua.q[0] = *(const uint4*)(arow + k0);
        ua.q[1] = *(const uint4*)(arow + k0 + 16);
        acc = WMMA_BF16(ua.v, ua.v, acc);
      }
#pragma unroll
      for (int r = 0; r < 8; ++r) G2[mhi + r][nl] = acc[r];
    }
    __syncthreads();

    // partial ||grad_W||^2 for this d-block: sum_{m,m'} (E E^T)_part * (C C^T)
    {
      const int ma = tid >> 4, mb = tid & 15;
      float g = 0.f;
      for (int dd = 0; dd < 64; ++dd) g += Esh[ma][dd] * Esh[mb][dd];
      red[tid] = g * G2[ma][mb];
    }
    __syncthreads();
    for (int w = 128; w > 0; w >>= 1) {
      if (tid < w) red[tid] += red[tid + w];
      __syncthreads();
    }
    const int slot = b * NCH + step;
    if (tid == 0) {
      gpart[(size_t)slot * NSPL + s] = red[0] * (1.0f / (MBK * MBK));
      __threadfence();
      atomicAdd(&gcnt[slot], 1);
    }
    // bias grad (local to this d-block), overlapped with the wait
    if (tid < 64) {
      float gb = 0.f;
#pragma unroll
      for (int m = 0; m < 16; ++m) gb += Esh[m][tid];
      gb *= (1.0f / MBK);
      gb = fminf(1.0f, fmaxf(-1.0f, gb));
      bvec[d0 + tid] -= LRC * gb;
    }
    if (tid == 0) {
      while (__hip_atomic_load(&gcnt[slot], __ATOMIC_ACQUIRE,
                               __HIP_MEMORY_SCOPE_AGENT) < NSPL) {
        __builtin_amdgcn_s_sleep(1);
      }
    }
    __syncthreads();
    float tot = 0.f;
#pragma unroll
    for (int i = 0; i < NSPL; ++i) tot += gpart[(size_t)slot * NSPL + i]; // deterministic order
    const float alpha = LRC / (MBK * fmaxf(sqrtf(tot), 1.0f));

    // Rank-16 fp32 update: W[d][e] -= alpha * sum_m E[m][d] * C[m][e].
    // Thread owns columns e0=tid, e1=tid+256 (coalesced row traffic, L2-resident W).
    {
      float c0[16], c1[16];
#pragma unroll
      for (int m = 0; m < 16; ++m) {
        c0[m] = bf2f(cz[m][tid]);
        c1[m] = bf2f(cz[m][tid + 256]);
      }
      for (int d = 0; d < 64; ++d) {
        float* wr = wblk + (size_t)d * HID;
        float w0 = wr[tid], w1 = wr[tid + 256];
#pragma unroll
        for (int m = 0; m < 16; ++m) {
          const float em = Esh[m][d];
          w0 -= alpha * em * c0[m];
          w1 -= alpha * em * c1[m];
        }
        wr[tid] = w0; wr[tid + 256] = w1;
      }
    }
  }
}

// ============ K3: y = x + preds @ W_out^T + b_out (pre-LN) ============
__global__ __launch_bounds__(256) void k3_out(
    const unsigned short* __restrict__ pbf, const unsigned short* __restrict__ woutbf,
    const float* __restrict__ bout, const float* __restrict__ x,
    float* __restrict__ y) {
  const int wave  = blockIdx.x * (blockDim.x >> 5) + (threadIdx.x >> 5);
  const int lane  = threadIdx.x & 31;
  const int mtile = wave >> 5;        // 1024 m-tiles
  const int strip = wave & 31;        // 32 strips of 64 columns over D
  const int nl  = lane & 15;
  const int kb  = (lane >> 4) << 3;
  const int mhi = (lane >> 4) << 3;
  const int n0  = strip << 6;

  const unsigned short* aptr = pbf + (size_t)((mtile << 4) + nl) * HID + kb;
  const unsigned short* b0 = woutbf + (size_t)(n0 + nl)      * HID + kb;
  const unsigned short* b1 = woutbf + (size_t)(n0 + 16 + nl) * HID + kb;
  const unsigned short* b2 = woutbf + (size_t)(n0 + 32 + nl) * HID + kb;
  const unsigned short* b3 = woutbf + (size_t)(n0 + 48 + nl) * HID + kb;

  v8f acc0 = {}, acc1 = {}, acc2 = {}, acc3 = {};
  v16bf a   = load_tile_bf(aptr);
  v16bf tb0 = load_tile_bf(b0);
  v16bf tb1 = load_tile_bf(b1);
  v16bf tb2 = load_tile_bf(b2);
  v16bf tb3 = load_tile_bf(b3);
  for (int k0 = 32; k0 < HID; k0 += 32) {
    v16bf an  = load_tile_bf(aptr + k0);
    v16bf bn0 = load_tile_bf(b0 + k0);
    v16bf bn1 = load_tile_bf(b1 + k0);
    v16bf bn2 = load_tile_bf(b2 + k0);
    v16bf bn3 = load_tile_bf(b3 + k0);
    __builtin_prefetch(aptr + k0 + 96, 0, 1);
    acc0 = WMMA_BF16(a, tb0, acc0);
    acc1 = WMMA_BF16(a, tb1, acc1);
    acc2 = WMMA_BF16(a, tb2, acc2);
    acc3 = WMMA_BF16(a, tb3, acc3);
    a = an; tb0 = bn0; tb1 = bn1; tb2 = bn2; tb3 = bn3;
  }
  acc0 = WMMA_BF16(a, tb0, acc0);
  acc1 = WMMA_BF16(a, tb1, acc1);
  acc2 = WMMA_BF16(a, tb2, acc2);
  acc3 = WMMA_BF16(a, tb3, acc3);

#pragma unroll
  for (int r = 0; r < 8; ++r) {
    const int mg = (mtile << 4) + mhi + r;
    v8f accs[4] = {acc0, acc1, acc2, acc3};
#pragma unroll
    for (int j = 0; j < 4; ++j) {
      const int ng = n0 + j * 16 + nl;
      const size_t o = (size_t)mg * DIM + ng;
      y[o] = accs[j][r] + bout[ng] + x[o];
    }
  }
}

// ============ K4: in-place LayerNorm over D ============
__global__ __launch_bounds__(256) void k4_ln(
    float* __restrict__ y, const float* __restrict__ g, const float* __restrict__ bb) {
  __shared__ float r1[256], r2[256];
  const int row = blockIdx.x;
  const int tid = threadIdx.x;
  float* yr = y + (size_t)row * DIM;
  float v[8], s = 0.f, s2 = 0.f;
#pragma unroll
  for (int i = 0; i < 8; ++i) {
    v[i] = yr[tid + i * 256];
    s += v[i]; s2 += v[i] * v[i];
  }
  r1[tid] = s; r2[tid] = s2;
  __syncthreads();
  for (int w = 128; w > 0; w >>= 1) {
    if (tid < w) { r1[tid] += r1[tid + w]; r2[tid] += r2[tid + w]; }
    __syncthreads();
  }
  const float mu  = r1[0] * (1.0f / DIM);
  const float var = r2[0] * (1.0f / DIM) - mu * mu;
  const float inv = rsqrtf(var + EPS);
#pragma unroll
  for (int i = 0; i < 8; ++i) {
    const int c = tid + i * 256;
    yr[c] = g[c] * (v[i] - mu) * inv + bb[c];
  }
}

// ================= host launcher =================
extern "C" void kernel_launch(void* const* d_in, const int* in_sizes, int n_in,
                              void* d_out, int out_size, void* d_ws, size_t ws_size,
                              hipStream_t stream) {
  const float* x      = (const float*)d_in[0];
  const float* W_in   = (const float*)d_in[1];
  const float* b_in   = (const float*)d_in[2];
  const float* W_tgt  = (const float*)d_in[3];
  const float* b_tgt  = (const float*)d_in[4];
  const float* W_out  = (const float*)d_in[5];
  const float* b_out  = (const float*)d_in[6];
  const float* W_init = (const float*)d_in[7];
  const float* b_init = (const float*)d_in[8];
  const float* ln_g   = (const float*)d_in[9];
  const float* ln_b   = (const float*)d_in[10];

  char* ws = (char*)d_ws;
  size_t off = 0;
  auto take = [&](size_t bytes) {
    char* p = ws + off;
    off = (off + bytes + 255) & ~(size_t)255;
    return p;
  };
  unsigned short* xbf    = (unsigned short*)take((size_t)BATCH * SEQ * DIM * 2);
  unsigned short* wzt    = (unsigned short*)take((size_t)2 * HID * DIM * 2);
  unsigned short* woutbf = (unsigned short*)take((size_t)DIM * HID * 2);
  unsigned short* zbf    = (unsigned short*)take((size_t)BATCH * SEQ * HID * 2);
  float*          tgtb   = (float*)         take((size_t)BATCH * SEQ * HID * 4);
  unsigned short* predsb = (unsigned short*)take((size_t)BATCH * SEQ * HID * 2);
  float*          Wst    = (float*)         take((size_t)BATCH * HID * HID * 4);
  float*          bst    = (float*)         take((size_t)BATCH * HID * 4);
  int*            gcnt   = (int*)           take((size_t)BATCH * NCH * 4);
  float*          gpart  = (float*)         take((size_t)BATCH * NCH * NSPL * 4);

  k0_setup<<<512, 256, 0, stream>>>(x, W_in, W_tgt, W_out, W_init, b_init,
                                    xbf, wzt, woutbf, Wst, bst, gcnt, gpart);
  // 1024 m-tiles * 16 strips = 16384 waves / 8 waves per block
  k1_proj<<<2048, 256, 0, stream>>>(xbf, wzt, b_in, b_tgt, zbf, tgtb);
  k2_scan<<<BATCH * NSPL, 256, 0, stream>>>(zbf, tgtb, Wst, bst, predsb, gcnt, gpart);
  // 1024 m-tiles * 32 strips = 32768 waves / 8 waves per block
  k3_out<<<4096, 256, 0, stream>>>(predsb, woutbf, b_out, x, (float*)d_out);
  k4_ln<<<BATCH * SEQ, 256, 0, stream>>>((float*)d_out, ln_g, ln_b);
}
